// GAUEncoder_27006754357257
// MI455X (gfx1250) — matrix operand order repaired and checked by reading, not compiled, not run on hardware
//
#include <hip/hip_runtime.h>
#include <hip/hip_bf16.h>

typedef __attribute__((ext_vector_type(16))) __bf16 v16bf;
typedef __attribute__((ext_vector_type(8)))  float  v8f;

#define BM 128
#define BN 128
#define BK 32
#define LSTR 40   // LDS row stride in bf16 elements (32 + 8 pad -> 80B rows, conflict-friendly)
#define LBUF (BM * LSTR)

__device__ __forceinline__ unsigned short f2bf(float f) {
    union { float f; unsigned u; } v; v.f = f;
    unsigned u = v.u;
    unsigned r = u + 0x7FFFu + ((u >> 16) & 1u);   // round-to-nearest-even
    return (unsigned short)(r >> 16);
}
__device__ __forceinline__ float bf2f(unsigned short h) {
    union { unsigned u; float f; } v; v.u = ((unsigned)h) << 16; return v.f;
}

// ---------------------------------------------------------------------------
// Shared NT GEMM core: C(128x128) += A(128xK) * B(128xK)^T, bf16 in, f32 acc.
// 8 waves: wave_m in 0..3 (32 rows each), wave_n in 0..1 (64 cols each);
// each wave computes 2x4 tiles of 16x16 via v_wmma_f32_16x16x32_bf16.
// Double-buffered LDS: one barrier per K-step; next tile staged (b128 loads)
// while WMMAs consume the current one.
// ---------------------------------------------------------------------------
__device__ __forceinline__
void gemm_tiles(const unsigned short* __restrict__ Ag, int lda,
                const unsigned short* __restrict__ Bg, int ldb,
                int K, unsigned short* As, unsigned short* Bs,   // each [2*LBUF]
                v8f (&acc)[2][4])
{
    const int tid  = threadIdx.x;
    const int lane = tid & 31;
    const int wm   = (tid >> 5) & 3;
    const int wn   = (tid >> 5) >> 2;
    const int col  = lane & 15;   // M (A frag) / N (B frag) index within 16-tile
    const int kh   = lane >> 4;   // K half selector

    // Staging indices: 512 uint4 (8 bf16) transfers per 128x32 tile, 2/thread
    const int sr0 = tid >> 2;                 // row 0..63
    const int sc0 = (tid & 3) << 3;           // col 0,8,16,24
    const int sr1 = sr0 + 64;

    v8f z = {};
    #pragma unroll
    for (int mt = 0; mt < 2; ++mt)
        #pragma unroll
        for (int nt = 0; nt < 4; ++nt) acc[mt][nt] = z;

    auto stage = [&](unsigned short* Asb, unsigned short* Bsb, int k0) {
        *(uint4*)(Asb + sr0 * LSTR + sc0) = *(const uint4*)(Ag + (size_t)sr0 * lda + k0 + sc0);
        *(uint4*)(Asb + sr1 * LSTR + sc0) = *(const uint4*)(Ag + (size_t)sr1 * lda + k0 + sc0);
        *(uint4*)(Bsb + sr0 * LSTR + sc0) = *(const uint4*)(Bg + (size_t)sr0 * ldb + k0 + sc0);
        *(uint4*)(Bsb + sr1 * LSTR + sc0) = *(const uint4*)(Bg + (size_t)sr1 * ldb + k0 + sc0);
    };

    stage(As, Bs, 0);
    __syncthreads();

    for (int k0 = 0; k0 < K; k0 += BK) {
        const int buf = (k0 / BK) & 1;
        unsigned short* Ac = As + buf * LBUF;
        unsigned short* Bc = Bs + buf * LBUF;
        if (k0 + BK < K)
            stage(As + (buf ^ 1) * LBUF, Bs + (buf ^ 1) * LBUF, k0 + BK);

        // A fragment (16x32): lane holds row 'col', K = {8*kh..8*kh+7} and +16
        v16bf afr[2], bfr[4];
        #pragma unroll
        for (int mt = 0; mt < 2; ++mt) {
            const unsigned short* p = Ac + (wm * 32 + mt * 16 + col) * LSTR + kh * 8;
            ((uint4*)&afr[mt])[0] = *(const uint4*)(p);
            ((uint4*)&afr[mt])[1] = *(const uint4*)(p + 16);
        }
        // B fragment (32x16): lane holds col 'col', K = {8*kh..8*kh+7} and +16
        #pragma unroll
        for (int nt = 0; nt < 4; ++nt) {
            const unsigned short* p = Bc + (wn * 64 + nt * 16 + col) * LSTR + kh * 8;
            ((uint4*)&bfr[nt])[0] = *(const uint4*)(p);
            ((uint4*)&bfr[nt])[1] = *(const uint4*)(p + 16);
        }
        #pragma unroll
        for (int mt = 0; mt < 2; ++mt)
            #pragma unroll
            for (int nt = 0; nt < 4; ++nt)
                acc[mt][nt] = __builtin_amdgcn_wmma_f32_16x16x32_bf16(
                    false, afr[mt], false, bfr[nt], (short)0, acc[mt][nt], false, false);

        // Publishes the staged (buf^1) tile AND retires this step's ds reads,
        // so (buf) may be overwritten next iteration. One barrier per K-step.
        __syncthreads();
    }
}

// Epilogue iterator: element (mt,nt,i) of acc maps to
//   mloc = wm*32 + mt*16 + (lane>>4)*8 + i,  nloc = wn*64 + nt*16 + (lane&15)
template <class F>
__device__ __forceinline__ void epilogue_foreach(v8f (&acc)[2][4], F f) {
    const int tid  = threadIdx.x;
    const int lane = tid & 31;
    const int wm   = (tid >> 5) & 3;
    const int wn   = (tid >> 5) >> 2;
    #pragma unroll
    for (int mt = 0; mt < 2; ++mt)
        #pragma unroll
        for (int nt = 0; nt < 4; ++nt)
            #pragma unroll
            for (int i = 0; i < 8; ++i) {
                int mloc = wm * 32 + mt * 16 + (lane >> 4) * 8 + i;
                int nloc = wn * 64 + nt * 16 + (lane & 15);
                f(mloc, nloc, acc[mt][nt][i]);
            }
}

// ---------------------------------------------------------------------------
// Kernel 0: fp32 -> bf16 weight conversion
// ---------------------------------------------------------------------------
__global__ void cvt_kernel(const float* __restrict__ in, unsigned short* __restrict__ out, int n) {
    int i = blockIdx.x * blockDim.x + threadIdx.x;
    if (i < n) out[i] = f2bf(in[i]);
}

// ---------------------------------------------------------------------------
// Kernel 1: row norm  xn = x * ln_g / max(||x||/sqrt(D), eps), bf16 out
// ---------------------------------------------------------------------------
__global__ void __launch_bounds__(256)
norm_kernel(const float* __restrict__ x, const float* __restrict__ ln_g,
            unsigned short* __restrict__ xn) {
    __shared__ float red[256];
    const int row = blockIdx.x;          // 0..8191
    const int tid = threadIdx.x;
    const float4 v = ((const float4*)(x + (size_t)row * 1024))[tid];
    red[tid] = v.x * v.x + v.y * v.y + v.z * v.z + v.w * v.w;
    __syncthreads();
    for (int s = 128; s > 0; s >>= 1) {
        if (tid < s) red[tid] += red[tid + s];
        __syncthreads();
    }
    const float norm  = sqrtf(red[0] * (1.0f / 1024.0f));
    const float scale = ln_g[0] / fmaxf(norm, 1e-5f);
    unsigned short* o = xn + (size_t)row * 1024 + tid * 4;
    o[0] = f2bf(v.x * scale); o[1] = f2bf(v.y * scale);
    o[2] = f2bf(v.z * scale); o[3] = f2bf(v.w * scale);
}

// ---------------------------------------------------------------------------
// Kernel 2: uv = silu(xn @ uv_w^T); scatter u, v^T, q, k
// grid = (33, 64), M=8192, N=4224, K=1024
// ---------------------------------------------------------------------------
__global__ void __launch_bounds__(256)
gau_gemm_uv(const unsigned short* __restrict__ xn, const unsigned short* __restrict__ w1,
            const float* __restrict__ pos_enc, const float* __restrict__ gamma,
            const float* __restrict__ beta,
            unsigned short* __restrict__ u, unsigned short* __restrict__ vT,
            unsigned short* __restrict__ qo, unsigned short* __restrict__ ko)
{
    __shared__ alignas(16) unsigned short As[2 * LBUF];
    __shared__ alignas(16) unsigned short Bs[2 * LBUF];
    const int bn = blockIdx.x, bm = blockIdx.y;
    v8f acc[2][4];
    gemm_tiles(xn + (size_t)bm * 128 * 1024, 1024,
               w1 + (size_t)bn * 128 * 1024, 1024, 1024, As, Bs, acc);
    epilogue_foreach(acc, [&](int mloc, int nloc, float cval) {
        int m = bm * 128 + mloc;                 // 0..8191 (= b*2048 + s)
        int n = bn * 128 + nloc;                 // 0..4223
        float sil = cval / (1.0f + __expf(-cval));
        int b = m >> 11;
        int s = m & 2047;
        if (n < 2048) {
            u[(size_t)m * 2048 + n] = f2bf(sil);
        } else if (n < 4096) {
            int e = n - 2048;                    // store v transposed: vT[b][e][s]
            vT[((size_t)b * 2048 + e) * 2048 + s] = f2bf(sil);
        } else {
            int ch = n - 4096;                   // 0..127
            float pe = pos_enc[(size_t)s * 128 + ch];
            qo[((size_t)b * 2048 + s) * 128 + ch] = f2bf(sil * gamma[ch]       + beta[ch]       + pe);
            ko[((size_t)b * 2048 + s) * 128 + ch] = f2bf(sil * gamma[128 + ch] + beta[128 + ch] + pe);
        }
    });
}

// ---------------------------------------------------------------------------
// Kernel 3: scores = relu(q @ k^T / sqrt(128))^2, per batch
// grid = (16, 16, 4), M=N=2048, K=128
// ---------------------------------------------------------------------------
__global__ void __launch_bounds__(256)
gau_gemm_scores(const unsigned short* __restrict__ q, const unsigned short* __restrict__ k,
                unsigned short* __restrict__ sc)
{
    __shared__ alignas(16) unsigned short As[2 * LBUF];
    __shared__ alignas(16) unsigned short Bs[2 * LBUF];
    const int bn = blockIdx.x, bm = blockIdx.y, bz = blockIdx.z;
    v8f acc[2][4];
    gemm_tiles(q + ((size_t)bz * 2048 + bm * 128) * 128, 128,
               k + ((size_t)bz * 2048 + bn * 128) * 128, 128, 128, As, Bs, acc);
    epilogue_foreach(acc, [&](int mloc, int nloc, float cval) {
        int m = bm * 128 + mloc;
        int n = bn * 128 + nloc;
        float sv = cval * 0.088388347648318447f;   // 1/sqrt(128)
        float r  = fmaxf(sv, 0.0f);
        sc[((size_t)bz * 2048 + m) * 2048 + n] = f2bf(r * r);
    });
}

// ---------------------------------------------------------------------------
// Kernel 4: g = u * (kernel @ v), per batch  (B = v^T so NT layout)
// grid = (16, 16, 4), M=2048, N=2048, K=2048
// ---------------------------------------------------------------------------
__global__ void __launch_bounds__(256)
gau_gemm_attnv(const unsigned short* __restrict__ sc, const unsigned short* __restrict__ vT,
               const unsigned short* __restrict__ u, unsigned short* __restrict__ g)
{
    __shared__ alignas(16) unsigned short As[2 * LBUF];
    __shared__ alignas(16) unsigned short Bs[2 * LBUF];
    const int bn = blockIdx.x, bm = blockIdx.y, bz = blockIdx.z;
    v8f acc[2][4];
    gemm_tiles(sc + ((size_t)bz * 2048 + bm * 128) * 2048, 2048,
               vT + ((size_t)bz * 2048 + bn * 128) * 2048, 2048, 2048, As, Bs, acc);
    epilogue_foreach(acc, [&](int mloc, int nloc, float cval) {
        size_t gi = ((size_t)bz * 2048 + bm * 128 + mloc) * 2048 + bn * 128 + nloc;
        g[gi] = f2bf(cval * bf2f(u[gi]));
    });
}

// ---------------------------------------------------------------------------
// Kernel 5: out = x * res_scale + g @ o_w^T
// grid = (8, 64), M=8192, N=1024, K=2048
// ---------------------------------------------------------------------------
__global__ void __launch_bounds__(256)
gau_gemm_out(const unsigned short* __restrict__ g, const unsigned short* __restrict__ ow,
             const float* __restrict__ x, const float* __restrict__ res_scale,
             float* __restrict__ out)
{
    __shared__ alignas(16) unsigned short As[2 * LBUF];
    __shared__ alignas(16) unsigned short Bs[2 * LBUF];
    const int bn = blockIdx.x, bm = blockIdx.y;
    v8f acc[2][4];
    gemm_tiles(g  + (size_t)bm * 128 * 2048, 2048,
               ow + (size_t)bn * 128 * 2048, 2048, 2048, As, Bs, acc);
    epilogue_foreach(acc, [&](int mloc, int nloc, float cval) {
        int m = bm * 128 + mloc;
        int n = bn * 128 + nloc;
        size_t oi = (size_t)m * 1024 + n;
        out[oi] = x[oi] * res_scale[n] + cval;
    });
}

// ---------------------------------------------------------------------------
extern "C" void kernel_launch(void* const* d_in, const int* in_sizes, int n_in,
                              void* d_out, int out_size, void* d_ws, size_t ws_size,
                              hipStream_t stream) {
    (void)in_sizes; (void)n_in; (void)out_size; (void)ws_size;
    const float* x     = (const float*)d_in[0];   // (4,2048,1024)
    const float* pos   = (const float*)d_in[1];   // (2048,128)
    const float* uvw   = (const float*)d_in[2];   // (4224,1024)
    const float* ow    = (const float*)d_in[3];   // (1024,2048)
    const float* gamma = (const float*)d_in[4];   // (2,128)
    const float* beta  = (const float*)d_in[5];   // (2,128)
    const float* lng   = (const float*)d_in[6];   // scalar
    const float* rs    = (const float*)d_in[7];   // (1024,)
    float* out = (float*)d_out;

    char* ws = (char*)d_ws;
    size_t off = 0;
    auto wsalloc = [&](size_t bytes) -> char* {
        char* p = ws + off;
        off += (bytes + 255) & ~(size_t)255;
        return p;
    };
    unsigned short* xn  = (unsigned short*)wsalloc((size_t)8192 * 1024 * 2);      // 16 MB
    unsigned short* w1  = (unsigned short*)wsalloc((size_t)4224 * 1024 * 2);      // 8.3 MB
    unsigned short* owb = (unsigned short*)wsalloc((size_t)1024 * 2048 * 2);      // 4 MB
    unsigned short* qb  = (unsigned short*)wsalloc((size_t)4 * 2048 * 128 * 2);   // 2 MB
    unsigned short* kb  = (unsigned short*)wsalloc((size_t)4 * 2048 * 128 * 2);   // 2 MB
    unsigned short* ub  = (unsigned short*)wsalloc((size_t)8192 * 2048 * 2);      // 32 MB
    unsigned short* vT  = (unsigned short*)wsalloc((size_t)4 * 2048 * 2048 * 2);  // 32 MB
    unsigned short* sc  = (unsigned short*)wsalloc((size_t)4 * 2048 * 2048 * 2);  // 32 MB
    unsigned short* gb  = (unsigned short*)wsalloc((size_t)8192 * 2048 * 2);      // 32 MB

    cvt_kernel<<<(4224 * 1024 + 255) / 256, 256, 0, stream>>>(uvw, w1, 4224 * 1024);
    cvt_kernel<<<(1024 * 2048 + 255) / 256, 256, 0, stream>>>(ow, owb, 1024 * 2048);
    norm_kernel<<<8192, 256, 0, stream>>>(x, lng, xn);
    gau_gemm_uv<<<dim3(33, 64), 256, 0, stream>>>(xn, w1, pos, gamma, beta, ub, vT, qb, kb);
    gau_gemm_scores<<<dim3(16, 16, 4), 256, 0, stream>>>(qb, kb, sc);
    gau_gemm_attnv<<<dim3(16, 16, 4), 256, 0, stream>>>(sc, vT, ub, gb);
    gau_gemm_out<<<dim3(8, 64), 256, 0, stream>>>(gb, owb, x, rs, out);
}
